// PointNetSetAbstraction_81990925681069
// MI455X (gfx1250) — matrix operand order repaired and checked by reading, not compiled, not run on hardware
//
#include <hip/hip_runtime.h>

typedef _Float16 v16h __attribute__((ext_vector_type(16)));
typedef float    v8f  __attribute__((ext_vector_type(8)));

// Problem constants (from reference setup_inputs / sa_forward defaults)
#define PB 16
#define PN 8192
#define PS 1024
#define PK 32
#define PD 64
#define PM ((long)PB * PS * PK)   // 524288 rows for the MLP GEMMs
#define RAD2 0.16f
#define BN_EPS 1e-5f

// ---------------------------------------------------------------------------
// 1) Farthest point sampling: one block per batch, 1024 threads, 8 pts/thread
// ---------------------------------------------------------------------------
__global__ __launch_bounds__(1024) void fps_kernel(const float* __restrict__ xyz,
                                                   int* __restrict__ fps_idx) {
  __shared__ float s_val[32];
  __shared__ int   s_idx[32];
  __shared__ float s_cent[3];

  const int b    = blockIdx.x;
  const int t    = threadIdx.x;
  const int wid  = t >> 5;
  const int lane = t & 31;

  float px[8], py[8], pz[8], dist[8];
#pragma unroll
  for (int i = 0; i < 8; ++i) {
    const int p = t + (i << 10);
    const float* q = xyz + ((long)b * PN + p) * 3;
    px[i] = q[0]; py[i] = q[1]; pz[i] = q[2];
    dist[i] = 1e10f;
  }

  int far = 0;
  for (int it = 0; it < PS; ++it) {
    if (t == 0) fps_idx[b * PS + it] = far;
    if (t == (far & 1023)) {
      const int sl = far >> 10;
      s_cent[0] = px[sl]; s_cent[1] = py[sl]; s_cent[2] = pz[sl];
    }
    __syncthreads();
    const float cx = s_cent[0], cy = s_cent[1], cz = s_cent[2];

    float bestv = -1.0f; int besti = 0;
#pragma unroll
    for (int i = 0; i < 8; ++i) {
      const float dx = px[i] - cx, dy = py[i] - cy, dz = pz[i] - cz;
      const float d = dx * dx + dy * dy + dz * dz;
      dist[i] = fminf(dist[i], d);
      const int p = t + (i << 10);
      if (dist[i] > bestv || (dist[i] == bestv && p < besti)) { bestv = dist[i]; besti = p; }
    }
    // wave32 argmax reduce (first-index tie break, matching jnp.argmax)
    for (int off = 16; off > 0; off >>= 1) {
      const float ov = __shfl_down(bestv, off, 32);
      const int   oi = __shfl_down(besti, off, 32);
      if (ov > bestv || (ov == bestv && oi < besti)) { bestv = ov; besti = oi; }
    }
    if (lane == 0) { s_val[wid] = bestv; s_idx[wid] = besti; }
    __syncthreads();
    if (wid == 0) {
      float v = s_val[lane]; int bi = s_idx[lane];
      for (int off = 16; off > 0; off >>= 1) {
        const float ov = __shfl_down(v, off, 32);
        const int   oi = __shfl_down(bi, off, 32);
        if (ov > v || (ov == v && oi < bi)) { v = ov; bi = oi; }
      }
      if (lane == 0) s_idx[0] = bi;
    }
    __syncthreads();
    far = s_idx[0];
  }
}

// ---------------------------------------------------------------------------
// 2) Gather new_xyz (also first output region)
// ---------------------------------------------------------------------------
__global__ void gather_newxyz(const float* __restrict__ xyz, const int* __restrict__ fps_idx,
                              float* __restrict__ new_xyz, float* __restrict__ out0) {
  const int q = blockIdx.x * blockDim.x + threadIdx.x;
  if (q >= PB * PS) return;
  const int b = q / PS;
  const int j = fps_idx[q];
  const float* src = xyz + ((long)b * PN + j) * 3;
  float* d0 = new_xyz + (long)q * 3;
  float* d1 = out0 + (long)q * 3;
  const float x = src[0], y = src[1], z = src[2];
  d0[0] = x; d0[1] = y; d0[2] = z;
  d1[0] = x; d1[1] = y; d1[2] = z;
}

// ---------------------------------------------------------------------------
// 3) query_ball_point: one wave32 per query, ordered compaction of first 32
// ---------------------------------------------------------------------------
__global__ __launch_bounds__(256) void query_ball(const float* __restrict__ xyz,
                                                  const float* __restrict__ new_xyz,
                                                  int* __restrict__ group_idx) {
  const int wave = threadIdx.x >> 5;
  const int lane = threadIdx.x & 31;
  const int q = blockIdx.x * 8 + wave;
  if (q >= PB * PS) return;
  const int b = q / PS;

  const float cx = new_xyz[(long)q * 3 + 0];
  const float cy = new_xyz[(long)q * 3 + 1];
  const float cz = new_xyz[(long)q * 3 + 2];

  const float* base = xyz + (long)b * PN * 3;
  int count = 0;
  int firstIdx = 0;

  for (int c = 0; c < PN; c += 32) {
    const int p = c + lane;
    const float dx = base[p * 3 + 0] - cx;
    const float dy = base[p * 3 + 1] - cy;
    const float dz = base[p * 3 + 2] - cz;
    const bool in = (dx * dx + dy * dy + dz * dz) <= RAD2;
    const unsigned mask = (unsigned)__ballot(in);
    if (count == 0 && mask != 0u) firstIdx = c + (__ffs(mask) - 1);
    if (in) {
      const int pos = count + __popc(mask & ((1u << lane) - 1u));
      if (pos < PK) group_idx[(long)q * PK + pos] = p;
    }
    count += __popc(mask);
    if (count >= PK) break;
  }
  if (count < PK) {
    for (int pos = count + lane; pos < PK; pos += 32)
      group_idx[(long)q * PK + pos] = firstIdx;
  }
}

// ---------------------------------------------------------------------------
// 4) Build feature rows: [dxyz(3) | points(64) | zero pad -> 96] as f16
// ---------------------------------------------------------------------------
__global__ void build_feats(const float* __restrict__ xyz, const float* __restrict__ points,
                            const float* __restrict__ new_xyz, const int* __restrict__ group_idx,
                            _Float16* __restrict__ X0) {
  const long row = (long)blockIdx.x * blockDim.x + threadIdx.x;
  if (row >= PM) return;
  const int b = (int)(row / ((long)PS * PK));
  const int s = (int)((row / PK) % PS);
  const int j = group_idx[row];

  const float* pj = xyz + ((long)b * PN + j) * 3;
  const float* cc = new_xyz + ((long)b * PS + s) * 3;
  const float* pf = points + ((long)b * PN + j) * PD;
  _Float16* dst = X0 + row * 96;

  dst[0] = (_Float16)(pj[0] - cc[0]);
  dst[1] = (_Float16)(pj[1] - cc[1]);
  dst[2] = (_Float16)(pj[2] - cc[2]);
#pragma unroll 8
  for (int i = 0; i < PD; ++i) dst[3 + i] = (_Float16)pf[i];
#pragma unroll
  for (int i = 67; i < 96; ++i) dst[i] = (_Float16)0.f;
}

// Convert/pad weights (O,Ci) f32 -> (O,KPAD) f16 with zero pad
__global__ void cvt_weights(const float* __restrict__ src, _Float16* __restrict__ dst,
                            int O, int Ci, int KPAD) {
  const int i = blockIdx.x * blockDim.x + threadIdx.x;
  if (i >= O * KPAD) return;
  const int o = i / KPAD, c = i % KPAD;
  dst[i] = (c < Ci) ? (_Float16)src[o * Ci + c] : (_Float16)0.f;
}

// ---------------------------------------------------------------------------
// 5) WMMA GEMM + bias + BN statistics.  Y[m,o] = sum_c X[m,c]*W[o,c] + b[o]
//    Block: 256 threads / 8 waves, tile 128 rows x O cols, K chunks of 32.
//    Tiles are staged in LDS *fragment-major*: frag[tile][lane][16 halves]
//    (lane stride padded to 24 halves = 48B, keeps 16B alignment for b128 DS
//    ops and spreads banks), so each lane's WMMA operand is two ds_load_b128.
// ---------------------------------------------------------------------------
union F16x16 {
  uint4 u[2];
  v16h  v;
};

template <int O, int KPAD>
__global__ __launch_bounds__(256) void gemm_bn(const _Float16* __restrict__ X,
                                               const _Float16* __restrict__ Wh,
                                               const float* __restrict__ bias,
                                               _Float16* __restrict__ Y,
                                               float* __restrict__ stats /*[256]*/) {
  constexpr int NT = O / 16;
  constexpr int KC = KPAD / 32;
  constexpr int LP = 24;  // lane stride in halves (48B): 16B-aligned, bank-spread

  __shared__ _Float16 aX[8][32][LP];   // A fragments for the 8 waves
  __shared__ _Float16 bW[NT][32][LP];  // B fragments for the NT column tiles
  __shared__ float sS1[O], sS2[O];

  const int t = threadIdx.x;
  const int wid = t >> 5, lane = t & 31;
  const int lo = lane & 15, hi = lane >> 4;
  const long mblock = (long)blockIdx.x * 128;

  if (t < O) { sS1[t] = 0.f; sS2[t] = 0.f; }

  v8f acc[NT];
#pragma unroll
  for (int nt = 0; nt < NT; ++nt)
#pragma unroll
    for (int r = 0; r < 8; ++r) acc[nt][r] = 0.f;

  for (int kc = 0; kc < KC; ++kc) {
    __syncthreads();
    // stage X tile (128 rows x 32 halves) directly into A-fragment layout.
    // Row r, k-group g (8 halves) -> wave r/16, lane (r%16)+16*(g&1),
    // fragment elements (g>>1)*8 .. +7  (per the 16-bit A-matrix VGPR map).
#pragma unroll
    for (int ld = 0; ld < 2; ++ld) {
      const int flat = t * 2 + ld;          // 0..511
      const int r = flat >> 2, g = flat & 3;
      const uint4 v = *(const uint4*)(X + (mblock + r) * KPAD + kc * 32 + g * 8);
      *(uint4*)&aX[r >> 4][(r & 15) + 16 * (g & 1)][(g >> 1) * 8] = v;
      if (kc + 1 < KC)  // gfx1250 global_prefetch_b8 for next K chunk
        __builtin_prefetch(X + (mblock + r) * KPAD + (kc + 1) * 32 + g * 8, 0, 1);
    }
    // stage W tile (O rows x 32 halves) into B-fragment layout.
    // W row n, k-group g -> tile n/16, lane (n%16)+16*(g&1), elems (g>>1)*8..
#pragma unroll
    for (int it = 0; it < (O * 4 + 255) / 256; ++it) {
      const int flat = it * 256 + t;
      if (flat < O * 4) {
        const int r = flat >> 2, g = flat & 3;
        const uint4 v = *(const uint4*)(Wh + (long)r * KPAD + kc * 32 + g * 8);
        *(uint4*)&bW[r >> 4][(r & 15) + 16 * (g & 1)][(g >> 1) * 8] = v;
      }
    }
    __syncthreads();

    // A fragment: two ds_load_b128 per lane
    F16x16 af;
    af.u[0] = *(const uint4*)&aX[wid][lane][0];
    af.u[1] = *(const uint4*)&aX[wid][lane][8];
#pragma unroll
    for (int nt = 0; nt < NT; ++nt) {
      F16x16 bf;
      bf.u[0] = *(const uint4*)&bW[nt][lane][0];
      bf.u[1] = *(const uint4*)&bW[nt][lane][8];
      acc[nt] = __builtin_amdgcn_wmma_f32_16x16x32_f16(
          false, af.v, false, bf.v, (short)0, acc[nt], false, false);
    }
  }

  // epilogue: bias, per-channel sums (f32), f16 store
#pragma unroll
  for (int nt = 0; nt < NT; ++nt) {
    const int n = nt * 16 + lo;
    const float bv = bias[n];
    float s1 = 0.f, s2 = 0.f;
#pragma unroll
    for (int r = 0; r < 8; ++r) {
      const float y = acc[nt][r] + bv;
      acc[nt][r] = y;
      s1 += y; s2 += y * y;
    }
    atomicAdd(&sS1[n], s1);
    atomicAdd(&sS2[n], s2);
#pragma unroll
    for (int r = 0; r < 8; ++r) {
      const long row = mblock + wid * 16 + r + 8 * hi;  // C layout: M = r + 8*hi
      Y[row * O + n] = (_Float16)acc[nt][r];
    }
  }
  __syncthreads();
  if (t < O) {
    atomicAdd(&stats[t], sS1[t]);
    atomicAdd(&stats[128 + t], sS2[t]);
  }
}

// mean/var -> per-channel scale/shift
__global__ void finalize_stats(const float* __restrict__ stats, const float* __restrict__ gamma,
                               const float* __restrict__ beta, float* __restrict__ scsh, int O) {
  const int o = threadIdx.x;
  if (o >= O) return;
  const float inv = 1.0f / (float)PM;
  const float mean = stats[o] * inv;
  const float var = stats[128 + o] * inv - mean * mean;
  const float sc = rsqrtf(var + BN_EPS) * gamma[o];
  scsh[o] = sc;
  scsh[128 + o] = beta[o] - mean * sc;
}

// in-place BN + ReLU on f16 activations
template <int O>
__global__ void normalize_relu(_Float16* __restrict__ X, const float* __restrict__ scsh, long total) {
  const long i = (long)blockIdx.x * blockDim.x + threadIdx.x;
  if (i >= total) return;
  const int o = (int)(i & (O - 1));
  float f = (float)X[i] * scsh[o] + scsh[128 + o];
  X[i] = (_Float16)(f > 0.f ? f : 0.f);
}

// final layer: BN + ReLU + max over nsample, write fp32 output region 2
__global__ __launch_bounds__(128) void norm_maxpool(const _Float16* __restrict__ Y3,
                                                    const float* __restrict__ scsh,
                                                    float* __restrict__ out1) {
  const int q = blockIdx.x;     // (b,s) flat: 16384
  const int o = threadIdx.x;    // 128 channels
  const float sc = scsh[o], sh = scsh[128 + o];
  float m = -3.4e38f;
#pragma unroll 4
  for (int k = 0; k < PK; ++k) {
    float f = (float)Y3[((long)q * PK + k) * 128 + o] * sc + sh;
    f = f > 0.f ? f : 0.f;
    m = fmaxf(m, f);
  }
  out1[(long)q * 128 + o] = m;
}

// ---------------------------------------------------------------------------
extern "C" void kernel_launch(void* const* d_in, const int* in_sizes, int n_in,
                              void* d_out, int out_size, void* d_ws, size_t ws_size,
                              hipStream_t stream) {
  (void)in_sizes; (void)n_in; (void)out_size; (void)ws_size;
  const float* xyz    = (const float*)d_in[0];
  const float* points = (const float*)d_in[1];
  const float* w[3]   = {(const float*)d_in[2], (const float*)d_in[6],  (const float*)d_in[10]};
  const float* bb[3]  = {(const float*)d_in[3], (const float*)d_in[7],  (const float*)d_in[11]};
  const float* gg[3]  = {(const float*)d_in[4], (const float*)d_in[8],  (const float*)d_in[12]};
  const float* bt[3]  = {(const float*)d_in[5], (const float*)d_in[9],  (const float*)d_in[13]};
  float* out = (float*)d_out;                 // [0,49152): new_xyz ; then new_points

  char* ws = (char*)d_ws;
  size_t off = 0;
  auto carve = [&](size_t bytes) { size_t o = off; off = (off + bytes + 255) & ~(size_t)255; return o; };
  int*      fps_idx   = (int*)(ws + carve((size_t)PB * PS * 4));
  float*    new_xyz   = (float*)(ws + carve((size_t)PB * PS * 3 * 4));
  int*      group_idx = (int*)(ws + carve((size_t)PB * PS * PK * 4));
  float*    stats     = (float*)(ws + carve(3 * 256 * 4));
  float*    scsh      = (float*)(ws + carve(3 * 256 * 4));
  _Float16* W0h       = (_Float16*)(ws + carve(64 * 96 * 2));
  _Float16* W1h       = (_Float16*)(ws + carve(64 * 64 * 2));
  _Float16* W2h       = (_Float16*)(ws + carve(128 * 64 * 2));
  _Float16* X0        = (_Float16*)(ws + carve((size_t)PM * 96 * 2));
  _Float16* X1        = (_Float16*)(ws + carve((size_t)PM * 64 * 2));
  _Float16* X2        = (_Float16*)(ws + carve((size_t)PM * 64 * 2));
  _Float16* Y3        = (_Float16*)(ws + carve((size_t)PM * 128 * 2));

  hipMemsetAsync(stats, 0, 3 * 256 * 4, stream);

  fps_kernel<<<PB, 1024, 0, stream>>>(xyz, fps_idx);
  gather_newxyz<<<(PB * PS + 255) / 256, 256, 0, stream>>>(xyz, fps_idx, new_xyz, out);
  query_ball<<<(PB * PS) / 8, 256, 0, stream>>>(xyz, new_xyz, group_idx);
  build_feats<<<(int)((PM + 255) / 256), 256, 0, stream>>>(xyz, points, new_xyz, group_idx, X0);

  cvt_weights<<<(64 * 96 + 255) / 256, 256, 0, stream>>>(w[0], W0h, 64, 67, 96);
  cvt_weights<<<(64 * 64 + 255) / 256, 256, 0, stream>>>(w[1], W1h, 64, 64, 64);
  cvt_weights<<<(128 * 64 + 255) / 256, 256, 0, stream>>>(w[2], W2h, 128, 64, 64);

  const int gBlocks = (int)(PM / 128);  // 4096

  gemm_bn<64, 96><<<gBlocks, 256, 0, stream>>>(X0, W0h, bb[0], X1, stats + 0);
  finalize_stats<<<1, 128, 0, stream>>>(stats + 0, gg[0], bt[0], scsh + 0, 64);
  normalize_relu<64><<<(int)(PM * 64 / 256), 256, 0, stream>>>(X1, scsh + 0, PM * 64);

  gemm_bn<64, 64><<<gBlocks, 256, 0, stream>>>(X1, W1h, bb[1], X2, stats + 256);
  finalize_stats<<<1, 128, 0, stream>>>(stats + 256, gg[1], bt[1], scsh + 256, 64);
  normalize_relu<64><<<(int)(PM * 64 / 256), 256, 0, stream>>>(X2, scsh + 256, PM * 64);

  gemm_bn<128, 64><<<gBlocks, 256, 0, stream>>>(X2, W2h, bb[2], Y3, stats + 512);
  finalize_stats<<<1, 128, 0, stream>>>(stats + 512, gg[2], bt[2], scsh + 512, 128);
  norm_maxpool<<<PB * PS, 128, 0, stream>>>(Y3, scsh + 512, out + (size_t)PB * PS * 3);
}
